// SelfAttentionV3_798863917455
// MI455X (gfx1250) — matrix operand order repaired and checked by reading, not compiled, not run on hardware
//
#include <hip/hip_runtime.h>

#define HDIM 128
#define SEQ  4096
#define NB   4
#define NKB  (SEQ / 32)
#define KPAD 136   // padded row (halfwords) for K tile in LDS
#define VPAD 40    // padded row for transposed V tile
#define PPAD 40    // padded row for P staging tile

typedef __attribute__((ext_vector_type(16))) unsigned short v16u;
typedef __attribute__((ext_vector_type(8)))  unsigned short v8u;
typedef __attribute__((ext_vector_type(16))) __bf16        v16bf;
typedef __attribute__((ext_vector_type(2)))  __bf16        v2bf;
typedef __attribute__((ext_vector_type(8)))  float         v8f;
typedef __attribute__((ext_vector_type(4)))  float         v4f;
typedef __attribute__((ext_vector_type(8)))  unsigned int  v8w;

union BFr { v16u u; v16bf b; v8w w; };

__device__ __forceinline__ unsigned short f2bf(float f) {
  unsigned u = __float_as_uint(f);
  u += 0x7FFFu + ((u >> 16) & 1u);       // round-to-nearest-even
  return (unsigned short)(u >> 16);
}

// Pack two f32 -> two bf16 in one DWORD (hardware cvt if the builtin exists)
__device__ __forceinline__ unsigned pk2(float lo, float hi) {
#if __has_builtin(__builtin_amdgcn_cvt_pk_bf16_f32)
  union { v2bf v; unsigned u; } t;
  t.v = __builtin_amdgcn_cvt_pk_bf16_f32(lo, hi);
  return t.u;
#else
  return (unsigned)f2bf(lo) | ((unsigned)f2bf(hi) << 16);
#endif
}

__device__ __forceinline__ v8f wmma_bf16(v16bf a, v16bf b, v8f c) {
  // D = A(16x32 bf16) * B(32x16 bf16) + C(16x16 f32)
  return __builtin_amdgcn_wmma_f32_16x16x32_bf16(false, a, false, b,
                                                 (short)0, c, false, false);
}

// Assemble a 16-halfword fragment from two 16B-aligned chunks.
__device__ __forceinline__ BFr ld_frag32(const unsigned short* p) {
  BFr r;
  v8u lo = *(const v8u*)p;
  v8u hi = *(const v8u*)(p + 8);
#pragma unroll
  for (int i = 0; i < 8; ++i) { r.u[i] = lo[i]; r.u[8 + i] = hi[i]; }
  return r;
}

// ---------------------------------------------------------------------------
// Kernel 0: one-shot W f32 -> bf16 (96 KB, stays L2-resident for kernel 1)
// ---------------------------------------------------------------------------
__global__ __launch_bounds__(256) void cvt_w_kernel(
    const float* __restrict__ W, unsigned short* __restrict__ Wb) {
  const int i = (blockIdx.x * 256 + threadIdx.x) * 8;
  v4f a = *(const v4f*)(W + i);
  v4f b = *(const v4f*)(W + i + 4);
  union { v8u u; unsigned w[4]; } r;
  r.w[0] = pk2(a[0], a[1]);
  r.w[1] = pk2(a[2], a[3]);
  r.w[2] = pk2(b[0], b[1]);
  r.w[3] = pk2(b[2], b[3]);
  *(v8u*)(Wb + i) = r.u;
}

// ---------------------------------------------------------------------------
// Kernel 1: fused QKV projection.  QKV[m][o] = sum_h X[m][h]*W[o][h] + b[o]
// One wave per 16-row M tile; processes TWO 16-col N tiles per iteration with
// independent accumulators so consecutive WMMAs alternate chains (no D->C
// hazard bubbles).  Writes Q (pre-scaled by 1/sqrt(128)), K, V as bf16.
// ---------------------------------------------------------------------------
__global__ __launch_bounds__(256) void qkv_proj_kernel(
    const float* __restrict__ X, const unsigned short* __restrict__ Wb,
    const float* __restrict__ bias,
    unsigned short* __restrict__ Qb, unsigned short* __restrict__ Kb,
    unsigned short* __restrict__ Vb) {
  const int lane = threadIdx.x & 31;
  const int hlf  = lane >> 4;
  const int lm   = lane & 15;
  const int wave = blockIdx.x * 8 + (threadIdx.x >> 5);
  const int m0   = wave * 16;

  // A fragments: X rows m0..m0+15, 4 chunks of K=32 (f32 -> bf16 inline)
  BFr afr[4];
  const float* xrow = X + (size_t)(m0 + lm) * HDIM;
#pragma unroll
  for (int c = 0; c < 4; ++c) {
    const float* p = xrow + 32 * c;
#pragma unroll
    for (int j = 0; j < 4; ++j)           // K = 16*hlf + 2j, 2j+1
      afr[c].w[j] = pk2(p[16 * hlf + 2 * j], p[16 * hlf + 2 * j + 1]);
#pragma unroll
    for (int j = 0; j < 4; ++j)           // K = 16*hlf + 8 + ...
      afr[c].w[4 + j] = pk2(p[16 * hlf + 8 + 2 * j], p[16 * hlf + 8 + 2 * j + 1]);
  }

  const float qscale = 0.08838834764831845f;  // 1/sqrt(128)
  const v8f vzero = {0.f, 0.f, 0.f, 0.f, 0.f, 0.f, 0.f, 0.f};

#pragma unroll 1
  for (int p = 0; p < 12; ++p) {              // pair of N tiles per iteration
    const int o0 = p * 32 + lm;               // output columns (per lane)
    const int o1 = o0 + 16;
    v8f acc0 = vzero, acc1 = vzero;
#pragma unroll
    for (int c = 0; c < 4; ++c) {
      // B fragment: element (K=h, N=o) = Wb[o][h]; contiguous 16 bf16 per lane
      BFr b0 = ld_frag32(&Wb[(size_t)o0 * HDIM + 32 * c + 16 * hlf]);
      BFr b1 = ld_frag32(&Wb[(size_t)o1 * HDIM + 32 * c + 16 * hlf]);
      acc0 = wmma_bf16(afr[c].b, b0.b, acc0);   // independent chains ->
      acc1 = wmma_bf16(afr[c].b, b1.b, acc1);   // back-to-back WMMA issue
    }
    const float bv0 = bias[o0];
    const float bv1 = bias[o1];
    unsigned short* dst;
    float scale = 1.0f;
    if (p < 4)      { dst = Qb; scale = qscale; }
    else if (p < 8) { dst = Kb; }
    else            { dst = Vb; }
    const int d0 = o0 & (HDIM - 1);
    const int d1 = o1 & (HDIM - 1);
#pragma unroll
    for (int v = 0; v < 8; v += 2) {          // D layout: row = v + 8*hlf
      const int m = m0 + v + 8 * hlf;
      const unsigned r0 = pk2((acc0[v] + bv0) * scale, (acc0[v + 1] + bv0) * scale);
      const unsigned r1 = pk2((acc1[v] + bv1) * scale, (acc1[v + 1] + bv1) * scale);
      dst[(size_t)m * HDIM + d0]       = (unsigned short)r0;
      dst[(size_t)(m + 1) * HDIM + d0] = (unsigned short)(r0 >> 16);
      dst[(size_t)m * HDIM + d1]       = (unsigned short)r1;
      dst[(size_t)(m + 1) * HDIM + d1] = (unsigned short)(r1 >> 16);
    }
  }
}

// ---------------------------------------------------------------------------
// Kernel 2: flash attention with double-buffered LDS + register prefetch.
// Block = 256 threads = 8 waves; each wave owns 16 query rows and a full
// 16x128 f32 output accumulator.  Per 32-key block: 8 QK^T WMMAs + online
// softmax + 8 PV WMMAs; next K/V tile's global loads are issued before the
// compute and drained into the alternate LDS buffer afterwards.
// ---------------------------------------------------------------------------
struct TileRegs { v8u k0, k1, w0, w1; };

__device__ __forceinline__ TileRegs load_tile(const unsigned short* Kp,
                                              const unsigned short* Vp,
                                              int kb, int t) {
  TileRegs r;
  const unsigned short* ks = Kp + (size_t)(kb * 32) * HDIM;
  r.k0 = *(const v8u*)&ks[t];
  r.k1 = *(const v8u*)&ks[t + 8];
  const unsigned short* vs = Vp + (size_t)(kb * 32) * HDIM;
  r.w0 = *(const v8u*)&vs[t];
  r.w1 = *(const v8u*)&vs[t + 8];
  return r;
}

__device__ __forceinline__ void store_tile(unsigned short* Kdst,
                                           unsigned short* Vdst,
                                           const TileRegs& r, int t) {
  const int row = t >> 7;          // key row 0..31
  const int c0  = t & (HDIM - 1);
  *(v8u*)&Kdst[row * KPAD + c0]     = r.k0;
  *(v8u*)&Kdst[row * KPAD + c0 + 8] = r.k1;
#pragma unroll
  for (int i = 0; i < 8; ++i) Vdst[(c0 + i) * VPAD + row]     = r.w0[i];
#pragma unroll
  for (int i = 0; i < 8; ++i) Vdst[(c0 + 8 + i) * VPAD + row] = r.w1[i];
}

__global__ __launch_bounds__(256) void flash_attn_kernel(
    const unsigned short* __restrict__ Qb,
    const unsigned short* __restrict__ Kb,
    const unsigned short* __restrict__ Vb,
    float* __restrict__ out) {
  __shared__ __align__(32) unsigned short Ksh[2][32 * KPAD];     // [key][d]
  __shared__ __align__(32) unsigned short Vsh[2][HDIM * VPAD];   // [d][key]
  __shared__ __align__(32) unsigned short Psh[8 * 16 * PPAD];    // per-wave P

  const int lane = threadIdx.x & 31;
  const int hlf  = lane >> 4;
  const int lm   = lane & 15;
  const int wid  = threadIdx.x >> 5;
  const int t    = threadIdx.x * 16;          // staging slice (16 halfwords)
  const int bat  = blockIdx.x >> 5;           // 32 q-blocks per batch
  const int q0   = (blockIdx.x & 31) * 128 + wid * 16;

  const unsigned short* Qp = Qb + (size_t)bat * SEQ * HDIM;
  const unsigned short* Kp = Kb + (size_t)bat * SEQ * HDIM;
  const unsigned short* Vp = Vb + (size_t)bat * SEQ * HDIM;

  // Q fragments (A layout), already scaled by 1/sqrt(d)
  BFr qf[4];
  const unsigned short* qrow = Qp + (size_t)(q0 + lm) * HDIM;
#pragma unroll
  for (int c = 0; c < 4; ++c) {
    const unsigned short* p = qrow + 32 * c;
#pragma unroll
    for (int i = 0; i < 8; ++i) qf[c].u[i]     = p[16 * hlf + i];
#pragma unroll
    for (int i = 0; i < 8; ++i) qf[c].u[8 + i] = p[16 * hlf + 8 + i];
  }

  v8f O[8];
  const v8f vzero = {0.f, 0.f, 0.f, 0.f, 0.f, 0.f, 0.f, 0.f};
#pragma unroll
  for (int dt = 0; dt < 8; ++dt) O[dt] = vzero;
  float mi[8], li[8];
#pragma unroll
  for (int v = 0; v < 8; ++v) { mi[v] = -1e30f; li[v] = 0.f; }

  unsigned short* pw = &Psh[wid * 16 * PPAD];

  // prologue: stage tile 0
  TileRegs tr = load_tile(Kp, Vp, 0, t);
  store_tile(Ksh[0], Vsh[0], tr, t);
  int cur = 0;

#pragma unroll 1
  for (int kb = 0; kb < NKB; ++kb) {
    __syncthreads();                          // buffer `cur` ready for all
    const bool more = (kb + 1) < NKB;
    if (more) tr = load_tile(Kp, Vp, kb + 1, t);   // issue early, drain later

    const unsigned short* Kcur = Ksh[cur];
    const unsigned short* Vcur = Vsh[cur];

    // S = Q K^T : two 16x16 score tiles (keys 0-15 and 16-31)
    v8f S0 = vzero, S1 = vzero;
#pragma unroll
    for (int c = 0; c < 4; ++c) {
      BFr b0 = ld_frag32(&Kcur[lm * KPAD + 32 * c + 16 * hlf]);
      BFr b1 = ld_frag32(&Kcur[(lm + 16) * KPAD + 32 * c + 16 * hlf]);
      S0 = wmma_bf16(qf[c].b, b0.b, S0);
      S1 = wmma_bf16(qf[c].b, b1.b, S1);
    }

    // online softmax (row stats live per lane-half, matching C-layout rows)
#pragma unroll
    for (int v = 0; v < 8; ++v) {
      float t0 = fmaxf(S0[v], S1[v]);
      t0 = fmaxf(t0, __shfl_xor(t0, 1, 32));
      t0 = fmaxf(t0, __shfl_xor(t0, 2, 32));
      t0 = fmaxf(t0, __shfl_xor(t0, 4, 32));
      t0 = fmaxf(t0, __shfl_xor(t0, 8, 32));
      const float mnew = fmaxf(mi[v], t0);
      const float p0   = __expf(S0[v] - mnew);
      const float p1   = __expf(S1[v] - mnew);
      const float corr = __expf(mi[v] - mnew);
      float rs = p0 + p1;
      rs += __shfl_xor(rs, 1, 32);
      rs += __shfl_xor(rs, 2, 32);
      rs += __shfl_xor(rs, 4, 32);
      rs += __shfl_xor(rs, 8, 32);
      li[v] = li[v] * corr + rs;
      mi[v] = mnew;
      S0[v] = p0;
      S1[v] = p1;
#pragma unroll
      for (int dt = 0; dt < 8; ++dt) O[dt][v] *= corr;
    }

    // stage P (D layout -> row-major bf16 16x32 in LDS)
#pragma unroll
    for (int v = 0; v < 8; ++v) {
      const int r = v + 8 * hlf;
      const unsigned pp = pk2(S0[v], S1[v]);
      pw[r * PPAD + lm]      = (unsigned short)pp;
      pw[r * PPAD + 16 + lm] = (unsigned short)(pp >> 16);
    }
    __builtin_amdgcn_wave_barrier();  // LDS is in-order per wave

    // reload P as A fragment (16x32)
    BFr pf;
    const unsigned short* prow = pw + lm * PPAD;
#pragma unroll
    for (int i = 0; i < 8; ++i) pf.u[i]     = prow[16 * hlf + i];
#pragma unroll
    for (int i = 0; i < 8; ++i) pf.u[8 + i] = prow[16 * hlf + 8 + i];

    // O += P * V  (8 d-tiles of 16 columns; independent chains)
#pragma unroll
    for (int dt = 0; dt < 8; ++dt) {
      BFr vf = ld_frag32(&Vcur[(dt * 16 + lm) * VPAD + 16 * hlf]);
      O[dt] = wmma_bf16(pf.b, vf.b, O[dt]);
    }

    // drain prefetched tile into the alternate buffer (safe: nobody reads
    // buffer cur^1 this iteration; next read is after the top barrier)
    if (more) store_tile(Ksh[cur ^ 1], Vsh[cur ^ 1], tr, t);
    cur ^= 1;
  }

  // epilogue: normalize by row sum and store f32
  float* op = out + (size_t)bat * SEQ * HDIM;
#pragma unroll
  for (int v = 0; v < 8; ++v) {
    const float inv = 1.0f / li[v];
    const int r = q0 + v + 8 * hlf;
#pragma unroll
    for (int dt = 0; dt < 8; ++dt)
      op[(size_t)r * HDIM + dt * 16 + lm] = O[dt][v] * inv;
  }
}

// ---------------------------------------------------------------------------
extern "C" void kernel_launch(void* const* d_in, const int* in_sizes, int n_in,
                              void* d_out, int out_size, void* d_ws, size_t ws_size,
                              hipStream_t stream) {
  (void)in_sizes; (void)n_in; (void)out_size; (void)ws_size;
  const float* X = (const float*)d_in[0];   // [4,4096,128]
  const float* W = (const float*)d_in[1];   // [384,128]
  const float* b = (const float*)d_in[2];   // [384]
  float* out = (float*)d_out;               // [4,4096,128]

  unsigned short* Qb = (unsigned short*)d_ws;          // bf16, 4 MB each
  unsigned short* Kb = Qb + (size_t)NB * SEQ * HDIM;
  unsigned short* Vb = Kb + (size_t)NB * SEQ * HDIM;
  unsigned short* Wb = Vb + (size_t)NB * SEQ * HDIM;   // bf16 W, 96 KB

  // 384*128 / (256*8) = 24 blocks
  cvt_w_kernel<<<dim3(24), dim3(256), 0, stream>>>(W, Wb);
  // 1024 M-tiles / 8 waves per block = 128 blocks
  qkv_proj_kernel<<<dim3((NB * SEQ / 16) / 8), dim3(256), 0, stream>>>(
      X, Wb, b, Qb, Kb, Vb);
  // 4 batches * 32 query blocks (128 queries each) = 128 blocks
  flash_attn_kernel<<<dim3(NB * (SEQ / 128)), dim3(256), 0, stream>>>(
      Qb, Kb, Vb, out);
}
